// Model_18296560681489
// MI455X (gfx1250) — matrix operand order repaired and checked by reading, not compiled, not run on hardware
//
#include <hip/hip_runtime.h>

typedef __bf16 bf16_t;
typedef __attribute__((ext_vector_type(16))) __bf16 v16bf;
typedef __attribute__((ext_vector_type(8)))  __bf16 v8bf;
typedef __attribute__((ext_vector_type(8)))  float  v8f;
typedef __attribute__((ext_vector_type(4)))  int    v4i;

typedef __attribute__((address_space(1))) v4i as1_v4i;
typedef __attribute__((address_space(3))) v4i as3_v4i;

#define BATCH 16384

// ---- CDNA5 async global->LDS copy (ASYNCcnt) with safe fallback ----------
#if defined(__has_builtin)
#if __has_builtin(__builtin_amdgcn_global_load_async_to_lds_b128)
#define HAVE_ASYNC_LDS 1
#endif
#endif

__device__ __forceinline__ void async_cp16(bf16_t* lds, const bf16_t* glob) {
#ifdef HAVE_ASYNC_LDS
  __builtin_amdgcn_global_load_async_to_lds_b128(
      (as1_v4i*)glob, (as3_v4i*)lds, 0, 0);
#else
  *(float4*)lds = *(const float4*)glob;
#endif
}

#ifdef HAVE_ASYNC_LDS
#define WAIT_ASYNC(n) asm volatile("s_wait_asynccnt " #n ::: "memory")
#else
#define WAIT_ASYNC(n) do {} while (0)
#endif

// ---------------------------------------------------------------------------
// Tiled WMMA GEMM:  C[M, ldc] = act( A[M,K] @ Bt[Npad,K]^T + bias )
// A, Bt bf16; fp32 accumulate via v_wmma_f32_16x16x32_bf16.
// Block tile 128x256, K-step 32, double-buffered async global->LDS staging.
// 256 threads = 8 waves (2x4); each wave computes 64x64 = 4x4 WMMA tiles.
// Steady-state loop is branch-free; last iteration peeled.
// ---------------------------------------------------------------------------
#define BM 128
#define BN 256
#define BK 32
#define LDSK 40   // padded LDS K-stride (bf16 elems): 80B rows, 16B aligned

template<bool RELU, bool OUT_BF16>
__global__ __launch_bounds__(256)
void gemm_bf16_wmma(const bf16_t* __restrict__ A,
                    const bf16_t* __restrict__ Bt,
                    const float*  __restrict__ bias,
                    void* __restrict__ Cout,
                    int K, int N_real, int ldc)
{
  __shared__ __align__(16) bf16_t As[2][BM * LDSK];   // 2 x 10 KB
  __shared__ __align__(16) bf16_t Bs[2][BN * LDSK];   // 2 x 20 KB

  const int tid     = threadIdx.x;
  const int lane    = tid & 31;
  const int wid     = tid >> 5;
  const int wm      = wid >> 2;     // 0..1 : 64 rows each
  const int wn      = wid & 3;      // 0..3 : 64 cols each
  const int laneRow = lane & 15;    // 0..15
  const int laneHi  = lane >> 4;    // 0 or 1

  const int bm = blockIdx.y * BM;
  const int bn = blockIdx.x * BN;

  v8f acc[4][4] = {};

  const int stR = tid >> 2;          // 0..63 staging row
  const int stC = (tid & 3) * 8;     // 0,8,16,24 (bf16 elems, 16B chunks)

  // stage one K-tile (6 x b128 async ops per thread)
  auto stage = [&](int buf, int k0) {
    #pragma unroll
    for (int p = 0; p < 2; ++p) {
      const int row = stR + p * 64;
      async_cp16(&As[buf][row * LDSK + stC],
                 A + (size_t)(bm + row) * K + k0 + stC);
    }
    #pragma unroll
    for (int p = 0; p < 4; ++p) {
      const int row = stR + p * 64;
      async_cp16(&Bs[buf][row * LDSK + stC],
                 Bt + (size_t)(bn + row) * K + k0 + stC);
    }
  };

  // 16 WMMAs against LDS buffer `cur`
  auto compute = [&](int cur) {
    v16bf bfrag[4];
    #pragma unroll
    for (int ni = 0; ni < 4; ++ni) {
      const bf16_t* p = &Bs[cur][(wn * 64 + ni * 16 + laneRow) * LDSK + laneHi * 16];
      v8bf b0 = *(const v8bf*)(p);
      v8bf b1 = *(const v8bf*)(p + 8);
      bfrag[ni] = __builtin_shufflevector(b0, b1,
                    0,1,2,3,4,5,6,7,8,9,10,11,12,13,14,15);
    }
    #pragma unroll
    for (int mi = 0; mi < 4; ++mi) {
      const bf16_t* p = &As[cur][(wm * 64 + mi * 16 + laneRow) * LDSK + laneHi * 16];
      v8bf a0 = *(const v8bf*)(p);
      v8bf a1 = *(const v8bf*)(p + 8);
      v16bf afrag = __builtin_shufflevector(a0, a1,
                      0,1,2,3,4,5,6,7,8,9,10,11,12,13,14,15);
      #pragma unroll
      for (int ni = 0; ni < 4; ++ni) {
        acc[mi][ni] = __builtin_amdgcn_wmma_f32_16x16x32_bf16(
            false, afrag, false, bfrag[ni], (short)0, acc[mi][ni], false, false);
      }
    }
  };

  const int nIter = K / BK;
  stage(0, 0);

  // steady state: branch-free (stage next, wait current, compute current)
  int it = 0;
  for (; it < nIter - 1; ++it) {
    const int cur = it & 1;
    stage(cur ^ 1, (it + 1) * BK);  // buffer cur^1 last read 2 barriers ago
    WAIT_ASYNC(6);                  // 6 in-flight belong to next tile
    __syncthreads();
    compute(cur);
    __syncthreads();
  }
  // peeled last iteration (no trailing barrier: epilogue is register-only)
  WAIT_ASYNC(0);
  __syncthreads();
  compute(it & 1);

  // ---- epilogue: bias + optional ReLU, store bf16 or f32 ----
  #pragma unroll
  for (int ni = 0; ni < 4; ++ni) {
    const int ncol = bn + wn * 64 + ni * 16 + laneRow;
    const float bv = (ncol < N_real) ? bias[ncol] : 0.0f;
    #pragma unroll
    for (int mi = 0; mi < 4; ++mi) {
      #pragma unroll
      for (int r = 0; r < 8; ++r) {
        const int mrow = bm + wm * 64 + mi * 16 + laneHi * 8 + r;
        float v = acc[mi][ni][r] + bv;
        if (RELU) v = fmaxf(v, 0.0f);
        if (ncol < ldc) {
          if (OUT_BF16)
            ((bf16_t*)Cout)[(size_t)mrow * ldc + ncol] = (bf16_t)v;
          else
            ((float*)Cout)[(size_t)mrow * ldc + ncol] = v;
        }
      }
    }
  }
}

// ---------------------------------------------------------------------------
// Weight prep: Wt[n*Kpad + k] = (k<K && n<N) ? W[k*N + n] : 0   (bf16)
// ---------------------------------------------------------------------------
__global__ void prep_wt(const float* __restrict__ W, bf16_t* __restrict__ Wt,
                        int K, int N, int Kpad, int Npad)
{
  const size_t idx = (size_t)blockIdx.x * blockDim.x + threadIdx.x;
  const size_t total = (size_t)Npad * Kpad;
  if (idx >= total) return;
  const int n = (int)(idx / Kpad);
  const int k = (int)(idx % Kpad);
  float v = (k < K && n < N) ? W[(size_t)k * N + n] : 0.0f;
  Wt[idx] = (bf16_t)v;
}

// Convert fp32 input [B,K] to zero-padded bf16 [B,Kpad]
__global__ void cvt_pad(const float* __restrict__ x, bf16_t* __restrict__ y,
                        int K, int Kpad)
{
  const size_t idx = (size_t)blockIdx.x * blockDim.x + threadIdx.x;
  if (idx >= (size_t)BATCH * Kpad) return;
  const int b = (int)(idx / Kpad);
  const int k = (int)(idx % Kpad);
  y[idx] = (k < K) ? (bf16_t)x[(size_t)b * K + k] : (bf16_t)0.0f;
}

// Zero the pad columns [610,640) of xcat once per launch
__global__ void zero_xcat_pad(bf16_t* __restrict__ xcat)
{
  const size_t idx = (size_t)blockIdx.x * blockDim.x + threadIdx.x;
  if (idx >= (size_t)BATCH * 30) return;
  const int b = (int)(idx / 30);
  const int j = (int)(idx % 30);
  xcat[(size_t)b * 640 + 610 + j] = (bf16_t)0.0f;
}

// loc head: h[b, 0..9] = lat*Wl[0,:] + lon*Wl[1,:] + bl
__global__ void loc_linear(const float* __restrict__ lat, const float* __restrict__ lon,
                           const float* __restrict__ Wl, const float* __restrict__ bl,
                           float* __restrict__ h, int ldh)
{
  const int b = blockIdx.x * blockDim.x + threadIdx.x;
  if (b >= BATCH) return;
  const float la = lat[b], lo = lon[b];
  #pragma unroll
  for (int j = 0; j < 10; ++j)
    h[(size_t)b * ldh + j] = la * Wl[j] + lo * Wl[10 + j] + bl[j];
}

// Per-column batch stats: mean[j], rstd[j] = rsqrt(var+eps) over B rows
__global__ void bn_stats(const float* __restrict__ h, int ldh,
                         float* __restrict__ mean, float* __restrict__ rstd)
{
  __shared__ float s1[256], s2[256];
  const int j = blockIdx.x;
  float a = 0.0f, bsum = 0.0f;
  for (int i = threadIdx.x; i < BATCH; i += 256) {
    float v = h[(size_t)i * ldh + j];
    a += v; bsum += v * v;
  }
  s1[threadIdx.x] = a; s2[threadIdx.x] = bsum;
  __syncthreads();
  for (int s = 128; s > 0; s >>= 1) {
    if (threadIdx.x < s) { s1[threadIdx.x] += s1[threadIdx.x + s];
                           s2[threadIdx.x] += s2[threadIdx.x + s]; }
    __syncthreads();
  }
  if (threadIdx.x == 0) {
    const float m = s1[0] / (float)BATCH;
    const float var = s2[0] / (float)BATCH - m * m;
    mean[j] = m;
    rstd[j] = rsqrtf(var + 1e-3f);
  }
}

// y = relu((h - m) * rstd * gamma + beta) -> bf16 into xcat[:, col0:col0+ncols]
__global__ void bn_apply(const float* __restrict__ h, int ldh, int ncols,
                         const float* __restrict__ mean, const float* __restrict__ rstd,
                         const float* __restrict__ gamma, const float* __restrict__ beta,
                         bf16_t* __restrict__ xcat, int ldx, int col0)
{
  const size_t idx = (size_t)blockIdx.x * blockDim.x + threadIdx.x;
  if (idx >= (size_t)BATCH * ncols) return;
  const int b = (int)(idx / ncols);
  const int j = (int)(idx % ncols);
  float v = (h[(size_t)b * ldh + j] - mean[j]) * rstd[j] * gamma[j] + beta[j];
  v = v > 0.0f ? v : 0.0f;
  xcat[(size_t)b * ldx + col0 + j] = (bf16_t)v;
}

// Final 512 -> 1: one wave per row, shuffle reduction
__global__ void final_dot(const bf16_t* __restrict__ act, const float* __restrict__ W7,
                          const float* __restrict__ b7, float* __restrict__ out, int K)
{
  const int row  = blockIdx.x * 8 + (threadIdx.x >> 5);
  const int lane = threadIdx.x & 31;
  if (row >= BATCH) return;
  float s = 0.0f;
  for (int k = lane; k < K; k += 32)
    s += (float)act[(size_t)row * K + k] * W7[k];
  #pragma unroll
  for (int off = 16; off > 0; off >>= 1)
    s += __shfl_down(s, off, 32);
  if (lane == 0) out[row] = s + b7[0];
}

// correct = sum(score_p - score_n > margin), single block, deterministic
__global__ void count_correct(const float* __restrict__ sp, const float* __restrict__ sn,
                              const int* __restrict__ margin, int* __restrict__ out)
{
  __shared__ int red[256];
  const float m = (float)margin[0];
  int cnt = 0;
  for (int i = threadIdx.x; i < BATCH; i += 256)
    if (sp[i] - sn[i] > m) cnt++;
  red[threadIdx.x] = cnt;
  __syncthreads();
  for (int s = 128; s > 0; s >>= 1) {
    if (threadIdx.x < s) red[threadIdx.x] += red[threadIdx.x + s];
    __syncthreads();
  }
  if (threadIdx.x == 0) out[0] = red[0];
}

// ---------------------------------------------------------------------------
extern "C" void kernel_launch(void* const* d_in, const int* in_sizes, int n_in,
                              void* d_out, int out_size, void* d_ws, size_t ws_size,
                              hipStream_t stream)
{
  (void)in_sizes; (void)n_in; (void)out_size; (void)ws_size;
  const int B = BATCH;

  const int*   margin = (const int*)  d_in[8];
  const float* Wi  = (const float*)d_in[9];   const float* bi  = (const float*)d_in[10];
  const float* gi  = (const float*)d_in[11];  const float* bti = (const float*)d_in[12];
  const float* Wtg = (const float*)d_in[13];  const float* btg = (const float*)d_in[14];
  const float* gt  = (const float*)d_in[15];  const float* btt = (const float*)d_in[16];
  const float* Wl  = (const float*)d_in[17];  const float* bl  = (const float*)d_in[18];
  const float* gl  = (const float*)d_in[19];  const float* btl = (const float*)d_in[20];
  const float* W1  = (const float*)d_in[21];  const float* b1  = (const float*)d_in[22];
  const float* W2  = (const float*)d_in[23];  const float* b2  = (const float*)d_in[24];
  const float* W3  = (const float*)d_in[25];  const float* b3  = (const float*)d_in[26];
  const float* W4  = (const float*)d_in[27];  const float* b4  = (const float*)d_in[28];
  const float* W5  = (const float*)d_in[29];  const float* b5  = (const float*)d_in[30];
  const float* W6  = (const float*)d_in[31];  const float* b6  = (const float*)d_in[32];
  const float* W7  = (const float*)d_in[33];  const float* b7  = (const float*)d_in[34];

  // -------- workspace carve --------
  char* ws = (char*)d_ws;
  size_t off = 0;
  auto carve = [&](size_t bytes) -> char* {
    off = (off + 255) & ~(size_t)255;
    char* p = ws + off; off += bytes; return p;
  };
  bf16_t* xcat = (bf16_t*)carve((size_t)B * 640 * 2);
  bf16_t* xin  = (bf16_t*)carve((size_t)B * 320 * 2);
  float*  hbuf = (float*) carve((size_t)B * 384 * 4);
  bf16_t* act1 = (bf16_t*)carve((size_t)B * 2048 * 2);
  bf16_t* act2 = (bf16_t*)carve((size_t)B * 2048 * 2);
  bf16_t* WtI  = (bf16_t*)carve((size_t)512 * 320 * 2);
  bf16_t* WtT  = (bf16_t*)carve((size_t)512 * 320 * 2);
  bf16_t* W1t  = (bf16_t*)carve((size_t)2048 * 640 * 2);
  bf16_t* W2t  = (bf16_t*)carve((size_t)2048 * 2048 * 2);
  bf16_t* W3t  = (bf16_t*)carve((size_t)1024 * 2048 * 2);
  bf16_t* W4t  = (bf16_t*)carve((size_t)1024 * 1024 * 2);
  bf16_t* W5t  = (bf16_t*)carve((size_t)512 * 1024 * 2);
  bf16_t* W6t  = (bf16_t*)carve((size_t)512 * 512 * 2);
  float*  meanb = (float*)carve(384 * 4);
  float*  rstdb = (float*)carve(384 * 4);

  auto cdiv = [](size_t a, size_t b) -> unsigned { return (unsigned)((a + b - 1) / b); };

  // -------- weight prep (transpose + pad + bf16) --------
  auto prep = [&](const float* W, bf16_t* Wt, int K, int N, int Kp, int Np) {
    prep_wt<<<cdiv((size_t)Np * Kp, 256), 256, 0, stream>>>(W, Wt, K, N, Kp, Np);
  };
  prep(Wi,  WtI, 300, 300, 320, 512);
  prep(Wtg, WtT, 300, 300, 320, 512);
  prep(W1,  W1t, 610, 2048, 640, 2048);
  prep(W2,  W2t, 2048, 2048, 2048, 2048);
  prep(W3,  W3t, 2048, 1024, 2048, 1024);
  prep(W4,  W4t, 1024, 1024, 1024, 1024);
  prep(W5,  W5t, 1024, 512, 1024, 512);
  prep(W6,  W6t, 512, 512, 512, 512);

  zero_xcat_pad<<<cdiv((size_t)B * 30, 256), 256, 0, stream>>>(xcat);

  for (int br = 0; br < 2; ++br) {
    const float* img = (const float*)d_in[br ? 4 : 0];
    const float* tag = (const float*)d_in[br ? 5 : 1];
    const float* lat = (const float*)d_in[br ? 6 : 2];
    const float* lon = (const float*)d_in[br ? 7 : 3];
    float* score = (float*)d_out + (br ? B : 0);

    // --- loc head: [B,2]x[2,10] + BN + ReLU -> xcat[:,0:10] ---
    loc_linear<<<cdiv(B, 256), 256, 0, stream>>>(lat, lon, Wl, bl, hbuf, 384);
    bn_stats<<<10, 256, 0, stream>>>(hbuf, 384, meanb, rstdb);
    bn_apply<<<cdiv((size_t)B * 10, 256), 256, 0, stream>>>(
        hbuf, 384, 10, meanb, rstdb, gl, btl, xcat, 640, 0);

    // --- img head: WMMA GEMM (K=320 padded) + BN + ReLU -> xcat[:,10:310] ---
    cvt_pad<<<cdiv((size_t)B * 320, 256), 256, 0, stream>>>(img, xin, 300, 320);
    gemm_bf16_wmma<false, false><<<dim3(2, B / BM), 256, 0, stream>>>(
        xin, WtI, bi, hbuf, 320, 300, 384);
    bn_stats<<<300, 256, 0, stream>>>(hbuf, 384, meanb, rstdb);
    bn_apply<<<cdiv((size_t)B * 300, 256), 256, 0, stream>>>(
        hbuf, 384, 300, meanb, rstdb, gi, bti, xcat, 640, 10);

    // --- tag head -> xcat[:,310:610] ---
    cvt_pad<<<cdiv((size_t)B * 320, 256), 256, 0, stream>>>(tag, xin, 300, 320);
    gemm_bf16_wmma<false, false><<<dim3(2, B / BM), 256, 0, stream>>>(
        xin, WtT, btg, hbuf, 320, 300, 384);
    bn_stats<<<300, 256, 0, stream>>>(hbuf, 384, meanb, rstdb);
    bn_apply<<<cdiv((size_t)B * 300, 256), 256, 0, stream>>>(
        hbuf, 384, 300, meanb, rstdb, gt, btt, xcat, 640, 310);

    // --- main MLP trunk (all WMMA, bf16 activations, fused bias+ReLU) ---
    gemm_bf16_wmma<true, true><<<dim3(8, B / BM), 256, 0, stream>>>(
        xcat, W1t, b1, act1, 640, 2048, 2048);
    gemm_bf16_wmma<true, true><<<dim3(8, B / BM), 256, 0, stream>>>(
        act1, W2t, b2, act2, 2048, 2048, 2048);
    gemm_bf16_wmma<true, true><<<dim3(4, B / BM), 256, 0, stream>>>(
        act2, W3t, b3, act1, 2048, 1024, 1024);
    gemm_bf16_wmma<true, true><<<dim3(4, B / BM), 256, 0, stream>>>(
        act1, W4t, b4, act2, 1024, 1024, 1024);
    gemm_bf16_wmma<true, true><<<dim3(2, B / BM), 256, 0, stream>>>(
        act2, W5t, b5, act1, 1024, 512, 512);
    gemm_bf16_wmma<true, true><<<dim3(2, B / BM), 256, 0, stream>>>(
        act1, W6t, b6, act2, 512, 512, 512);

    final_dot<<<cdiv(B, 8), 256, 0, stream>>>(act2, W7, b7, score, 512);
  }

  count_correct<<<1, 256, 0, stream>>>(
      (const float*)d_out, (const float*)d_out + B, margin,
      (int*)((float*)d_out + 2 * B));
}
